// ResidualBlockNoBN_DynaMic_72550587564279
// MI455X (gfx1250) — compile-verified, matrix operand order checked by reading, not run on hardware
//
#include <hip/hip_runtime.h>
#include <hip/hip_bf16.h>

// ---------------- problem constants ----------------
#define BATCH 16
#define CCH   64
#define HH    192
#define WW    192
#define HWPIX (HH*WW)          // 36864
#define NEXP  7

// ---------------- workspace layout (bytes) ----------------
// gate:  16*7 f32                      @ 0
// kb1:   16*64 f32                     @ 4096
// kb2:   16*64 f32                     @ 8192
// ap1:   16*36864 bf16 (packed A)      @ 16384        (1,179,648 B)
// ap2:   16*36864 bf16                 @ 1,196,032    (1,179,648 B)
// out1:  16*64*192*192 bf16            @ 2,375,680    (75,497,472 B)
#define GATE_OFF 0
#define KB1_OFF  4096
#define KB2_OFF  8192
#define AP1_OFF  16384
#define AP2_OFF  1196032
#define OUT1_OFF 2375680

typedef __attribute__((ext_vector_type(16))) __bf16          v16bf;
typedef __attribute__((ext_vector_type(8)))  float           v8f;
typedef __attribute__((ext_vector_type(8)))  unsigned short  su8;   // 16 B
typedef __attribute__((ext_vector_type(16))) unsigned short  su16;  // 32 B

union BFrag { su16 u; v16bf b; };

__device__ __forceinline__ unsigned short f2bf(float f) {
    unsigned int u = __builtin_bit_cast(unsigned int, f);
    unsigned int r = u + 0x7FFFu + ((u >> 16) & 1u);   // round-to-nearest-even
    return (unsigned short)(r >> 16);
}

// ------------------------------------------------------------------
// Kernel 1: gate = softmax( mean_HW(x) @ gate_w^T + gate_b )   [16][7]
// ------------------------------------------------------------------
__global__ __launch_bounds__(256) void gate_kernel(
    const float* __restrict__ x, const float* __restrict__ gw,
    const float* __restrict__ gb, float* __restrict__ gate)
{
    __shared__ float part[256];
    __shared__ float pooled[CCH];
    __shared__ float logits[NEXP];
    const int b = blockIdx.x, tid = threadIdx.x;
    const int c = tid >> 2, q = tid & 3;
    const float* p = x + ((long)b * CCH + c) * HWPIX;
    float s = 0.f;
    for (int i = 0; i < HWPIX / 4; ++i) s += p[q + 4 * i];
    part[tid] = s;
    __syncthreads();
    if (tid < CCH)
        pooled[tid] = (part[tid*4] + part[tid*4+1] + part[tid*4+2] + part[tid*4+3])
                      * (1.0f / (float)HWPIX);
    __syncthreads();
    if (tid < NEXP) {
        float l = gb[tid];
        for (int cc = 0; cc < CCH; ++cc) l += pooled[cc] * gw[tid * CCH + cc];
        logits[tid] = l;
    }
    __syncthreads();
    if (tid == 0) {
        float m = logits[0];
        for (int e = 1; e < NEXP; ++e) m = fmaxf(m, logits[e]);
        float ex[NEXP], den = 0.f;
        for (int e = 0; e < NEXP; ++e) { ex[e] = __expf(logits[e] - m); den += ex[e]; }
        for (int e = 0; e < NEXP; ++e) gate[b * NEXP + e] = ex[e] / den;
    }
}

// ------------------------------------------------------------------
// Kernel 2: mix weights with gate, bf16-convert, and pre-swizzle into
// the per-lane WMMA A-fragment layout (ISA 16-bit A 16x32 table):
//   lane l: m = l%16, half = l/16
//   frag element e (0..15): v = e/2, kbase = (v<4?0:16)+(half?8:0)+2*(v&3)
//   K index = ks*32 + kbase + (e&1)
// frag id f = ((tap*4 + mtile)*2 + ks), element addr = f*512 + lane*16 + e
// Also mixes biases: kb[o] = conv_b[o] + sum_e g[e]*expert_b[e][o]
// ------------------------------------------------------------------
__global__ __launch_bounds__(256) void pack_kernel(
    const float* __restrict__ cw, const float* __restrict__ cb,
    const float* __restrict__ ew, const float* __restrict__ eb,
    const float* __restrict__ gate,
    unsigned short* __restrict__ apack, float* __restrict__ kb)
{
    __shared__ float g[NEXP];
    const int b = blockIdx.x, tid = threadIdx.x;
    if (tid < NEXP) g[tid] = gate[b * NEXP + tid];
    __syncthreads();
    if (tid < CCH) {
        float v = cb[tid];
        #pragma unroll
        for (int e = 0; e < NEXP; ++e) v += g[e] * eb[e * CCH + tid];
        kb[b * CCH + tid] = v;
    }
    unsigned short* ab = apack + (long)b * (CCH * CCH * 9);
    for (int idx = tid; idx < CCH * CCH * 9; idx += 256) {
        const int e16  = idx & 15;
        const int lane = (idx >> 4) & 31;
        const int frag = idx >> 9;
        const int ks   = frag & 1;
        const int mt   = (frag >> 1) & 3;
        const int tap  = frag >> 3;
        const int half = lane >> 4;
        const int o    = mt * 16 + (lane & 15);
        const int v    = e16 >> 1;
        const int kbase = (v < 4 ? 0 : 16) + (half ? 8 : 0) + 2 * (v & 3);
        const int i    = ks * 32 + kbase + (e16 & 1);
        const long widx = ((long)o * CCH + i) * 9 + tap;
        float wv = cw[widx];
        #pragma unroll
        for (int e = 0; e < NEXP; ++e)
            wv += g[e] * ew[(((long)e * CCH + o) * CCH + i) * 9 + tap];
        ab[idx] = f2bf(wv);
    }
}

// ------------------------------------------------------------------
// Conv kernels: one block per (sample, output row). 8 waves:
//   mtile = wave%4 (16 out channels), nhalf = wave/4 (6 of 12 col tiles).
// LDS staged channel-contiguous [3 rows][194 cols][32 ch] bf16 per K phase,
// so a B fragment is two ds_load_b128 per lane.
// Per tap: batch-load all 6 B fragments (12 ds_load_b128 into distinct
// registers) BEFORE the 6 WMMAs so the scheduler can overlap LDS latency
// with the matrix pipe instead of wait-dscnt-0 per WMMA.
// ------------------------------------------------------------------
__device__ __forceinline__ void conv_core(
    v8f acc[6], const unsigned short* __restrict__ smem,
    const su16* __restrict__ ap, int mt, int nh, int lane, int ks)
{
    const int n = lane & 15, half = lane >> 4;
    #pragma unroll
    for (int tap = 0; tap < 9; ++tap) {
        const int kh = tap / 3, kw = tap % 3;
        BFrag a;
        a.u = ap[(((tap * 4 + mt) * 2 + ks) * 32) + lane];   // 32B coalesced
        BFrag bb[6];
        #pragma unroll
        for (int j = 0; j < 6; ++j) {
            const int col = (nh * 6 + j) * 16 + n + kw;       // 0..193
            const unsigned short* base = smem + (kh * 194 + col) * 32 + half * 8;
            union { su8 h[2]; su16 v; } t;
            t.h[0] = *(const su8*)(base);        // ch kbase 0..7   (ds_load_b128)
            t.h[1] = *(const su8*)(base + 16);   // ch kbase 16..23 (ds_load_b128)
            bb[j].u = t.v;
        }
        #pragma unroll
        for (int j = 0; j < 6; ++j)
            acc[j] = __builtin_amdgcn_wmma_f32_16x16x32_bf16(
                false, a.b, false, bb[j].b, (short)0, acc[j], false, false);
    }
}

__global__ __launch_bounds__(256) void conv1_kernel(
    const float* __restrict__ x, const unsigned short* __restrict__ apack,
    const float* __restrict__ kb, unsigned short* __restrict__ out1)
{
    __shared__ unsigned short smem[3 * 194 * 32];    // 37,248 B per phase
    const int b = blockIdx.x / HH, y = blockIdx.x % HH;
    const int tid = threadIdx.x;
    const int wave = tid >> 5, lane = tid & 31;
    const int mt = wave & 3, nh = wave >> 2;

    v8f acc[6];
    #pragma unroll
    for (int j = 0; j < 6; ++j)
        #pragma unroll
        for (int r = 0; r < 8; ++r) acc[j][r] = 0.f;

    const su16* ap = (const su16*)(apack + (long)b * (CCH * CCH * 9));

    for (int ks = 0; ks < 2; ++ks) {
        // stage channels [ks*32, ks*32+32) of rows y-1..y+1 as bf16, ch-contiguous
        for (int p = tid; p < 3 * 194; p += 256) {
            const int r = p / 194, cl = p % 194;
            const int iny = y + r - 1, inx = cl - 1;
            const bool valid = (iny >= 0) & (iny < HH) & (inx >= 0) & (inx < WW);
            const float* src = x + (long)b * CCH * HWPIX + (long)iny * WW + inx;
            unsigned short* dst = smem + p * 32;
            __builtin_prefetch(src + WW, 0, 0);
            #pragma unroll
            for (int c0 = 0; c0 < 32; c0 += 8) {
                su8 v;
                #pragma unroll
                for (int c = 0; c < 8; ++c) {
                    const float f = valid ? src[(long)(ks * 32 + c0 + c) * HWPIX] : 0.f;
                    v[c] = f2bf(f);
                }
                *(su8*)(dst + c0) = v;           // ds_store_b128
            }
        }
        __syncthreads();
        conv_core(acc, smem, ap, mt, nh, lane, ks);
        __syncthreads();
    }

    // epilogue: bias + ReLU, store bf16
    const float* bias = kb + b * CCH;
    const int n = lane & 15, half = lane >> 4;
    #pragma unroll
    for (int j = 0; j < 6; ++j) {
        const int px = (nh * 6 + j) * 16 + n;
        #pragma unroll
        for (int r = 0; r < 8; ++r) {
            const int o = mt * 16 + r + half * 8;
            float v = acc[j][r] + bias[o];
            v = v > 0.f ? v : 0.f;
            out1[(((long)b * CCH + o) * HH + y) * WW + px] = f2bf(v);
        }
    }
}

__global__ __launch_bounds__(256) void conv2_kernel(
    const unsigned short* __restrict__ in1, const unsigned short* __restrict__ apack,
    const float* __restrict__ kb, const float* __restrict__ x,
    float* __restrict__ out)
{
    __shared__ unsigned short smem[3 * 194 * 32];
    const int b = blockIdx.x / HH, y = blockIdx.x % HH;
    const int tid = threadIdx.x;
    const int wave = tid >> 5, lane = tid & 31;
    const int mt = wave & 3, nh = wave >> 2;

    v8f acc[6];
    #pragma unroll
    for (int j = 0; j < 6; ++j)
        #pragma unroll
        for (int r = 0; r < 8; ++r) acc[j][r] = 0.f;

    const su16* ap = (const su16*)(apack + (long)b * (CCH * CCH * 9));

    for (int ks = 0; ks < 2; ++ks) {
        for (int p = tid; p < 3 * 194; p += 256) {
            const int r = p / 194, cl = p % 194;
            const int iny = y + r - 1, inx = cl - 1;
            const bool valid = (iny >= 0) & (iny < HH) & (inx >= 0) & (inx < WW);
            const unsigned short* src = in1 + (long)b * CCH * HWPIX + (long)iny * WW + inx;
            unsigned short* dst = smem + p * 32;
            #pragma unroll
            for (int c0 = 0; c0 < 32; c0 += 8) {
                su8 v;
                #pragma unroll
                for (int c = 0; c < 8; ++c)
                    v[c] = valid ? src[(long)(ks * 32 + c0 + c) * HWPIX] : (unsigned short)0;
                *(su8*)(dst + c0) = v;
            }
        }
        __syncthreads();
        conv_core(acc, smem, ap, mt, nh, lane, ks);
        __syncthreads();
    }

    // epilogue: bias + residual (RES_SCALE = 1.0), fp32 out
    const float* bias = kb + b * CCH;
    const int n = lane & 15, half = lane >> 4;
    #pragma unroll
    for (int j = 0; j < 6; ++j) {
        const int px = (nh * 6 + j) * 16 + n;
        #pragma unroll
        for (int r = 0; r < 8; ++r) {
            const int o = mt * 16 + r + half * 8;
            const long oi = (((long)b * CCH + o) * HH + y) * WW + px;
            out[oi] = x[oi] + (acc[j][r] + bias[o]);
        }
    }
}

// ------------------------------------------------------------------
extern "C" void kernel_launch(void* const* d_in, const int* in_sizes, int n_in,
                              void* d_out, int out_size, void* d_ws, size_t ws_size,
                              hipStream_t stream) {
    const float* x   = (const float*)d_in[0];
    const float* c1w = (const float*)d_in[1];
    const float* c1b = (const float*)d_in[2];
    const float* c2w = (const float*)d_in[3];
    const float* c2b = (const float*)d_in[4];
    const float* e1w = (const float*)d_in[5];
    const float* e1b = (const float*)d_in[6];
    const float* e2w = (const float*)d_in[7];
    const float* e2b = (const float*)d_in[8];
    const float* gw  = (const float*)d_in[9];
    const float* gb  = (const float*)d_in[10];

    char* ws = (char*)d_ws;
    float*          gate = (float*)(ws + GATE_OFF);
    float*          kb1  = (float*)(ws + KB1_OFF);
    float*          kb2  = (float*)(ws + KB2_OFF);
    unsigned short* ap1  = (unsigned short*)(ws + AP1_OFF);
    unsigned short* ap2  = (unsigned short*)(ws + AP2_OFF);
    unsigned short* out1 = (unsigned short*)(ws + OUT1_OFF);   // needs ~78 MB ws

    gate_kernel<<<BATCH, 256, 0, stream>>>(x, gw, gb, gate);
    pack_kernel<<<BATCH, 256, 0, stream>>>(c1w, c1b, e1w, e1b, gate, ap1, kb1);
    pack_kernel<<<BATCH, 256, 0, stream>>>(c2w, c2b, e2w, e2b, gate, ap2, kb2);
    conv1_kernel<<<BATCH * HH, 256, 0, stream>>>(x, ap1, kb1, out1);
    conv2_kernel<<<BATCH * HH, 256, 0, stream>>>(out1, ap2, kb2, x, (float*)d_out);
}
